// GraphNet_72541997629573
// MI455X (gfx1250) — compile-verified, matrix operand order checked by reading, not run on hardware
//
#include <hip/hip_runtime.h>
#include <hip/hip_bf16.h>
#include <math.h>

// ---------------------------------------------------------------------------
// GATv2 forward for MI455X (gfx1250, wave32, WMMA + TDM).
//   N=16384 nodes, E=131072 edges, HID=256, H=8 heads x 32 ch, OUT=128.
// Dense projections: v_wmma_f32_16x16x32_bf16 (f32 accum), weights repacked
// into B-fragment layout, staged into LDS per block via tensor_load_to_lds
// (TDM, s_wait_tensorcnt), then consumed with ds_load_b128.
// The e@We GEMM is fused into the edge attention-logit kernel.
// ---------------------------------------------------------------------------

typedef __attribute__((ext_vector_type(16))) __bf16      v16bf;
typedef __attribute__((ext_vector_type(8)))  float       v8f;
typedef __attribute__((ext_vector_type(4)))  unsigned    u32x4;
typedef __attribute__((ext_vector_type(8)))  int         i32x8;
typedef __attribute__((ext_vector_type(4)))  int         i32x4;
typedef __attribute__((ext_vector_type(4)))  float       f32x4;

#define HIDD   256
#define NNODES 16384
#define NEDGES 131072
#define NHEAD  8
#define OUTD   128
#define NEG_SLOPE 0.2f

// ---- float atomic max via signed/unsigned integer ordering trick ----------
__device__ __forceinline__ void atomicMaxF32(float* addr, float v) {
  if (v >= 0.0f) atomicMax((int*)addr, __float_as_int(v));
  else           atomicMin((unsigned int*)addr, __float_as_uint(v));
}

// ---------------------------------------------------------------------------
// TDM: 1-D bulk copy global -> LDS (D# per CDNA5 ISA).
//   data_size = 8 bytes, tensor_dim0 = tile_dim0 = nbytes/8, 1-D tile.
//   group0: count=1 | lds_addr | global_addr[56:0] | type=2.
// Issued by one wave (TDM ignores EXEC); completion via s_wait_tensorcnt.
// ---------------------------------------------------------------------------
__device__ __forceinline__ void tdm_load_1d(unsigned lds_off, const void* gptr,
                                            unsigned nbytes) {
  unsigned long long ga = (unsigned long long)(uintptr_t)gptr;
  unsigned units = nbytes >> 3;                        // 8-byte elements
  u32x4 g0;
  g0[0] = 1u;                                          // count=1, user mode
  g0[1] = lds_off;                                     // lds_addr (bytes)
  g0[2] = (unsigned)ga;                                // global_addr[31:0]
  g0[3] = (unsigned)((ga >> 32) & 0x01ffffffull)       // global_addr[56:32]
        | (2u << 30);                                  // type = 2 ("image")
  i32x8 g1;
  g1[0] = (int)(3u << 16);                             // data_size=8B, mask=0
  g1[1] = (int)((units & 0xffffu) << 16);              // tensor_dim0 lo @63:48
  g1[2] = (int)(((units >> 16) & 0xffffu)              // tensor_dim0 hi @79:64
        | (1u << 16));                                 // tensor_dim1=1 @111:80
  g1[3] = (int)((units & 0xffffu) << 16);              // tile_dim0 @127:112
  g1[4] = 0;                                           // tile_dim1/2 unused
  g1[5] = (int)units;                                  // dim0_stride @191:160
  g1[6] = 0;
  g1[7] = 0;
  i32x4 z4 = {0, 0, 0, 0};
#if defined(__clang_major__) && (__clang_major__ >= 23)
  i32x8 z8 = {0, 0, 0, 0, 0, 0, 0, 0};
  __builtin_amdgcn_tensor_load_to_lds(g0, g1, z4, z4, z8, 0);
#else
  __builtin_amdgcn_tensor_load_to_lds(g0, g1, z4, z4, 0);
#endif
}

// ---------------------------------------------------------------------------
// Repack f32 weight [K=256][ncols] into bf16 WMMA B-fragments.
// Lane = nn + 16*((kk>>3)&1); slot = 2*j + (kk&1), j = ((kk&7)>>1)+4*((kk>>4)&1).
// Tile order: tile = nb*(K/32) + kb.
// ---------------------------------------------------------------------------
__global__ void repack_w(const float* __restrict__ W, __bf16* __restrict__ P,
                         int ncols) {
  int idx = blockIdx.x * blockDim.x + threadIdx.x;
  int tot = HIDD * ncols;
  if (idx >= tot) return;
  int k = idx / ncols, n = idx - k * ncols;
  int kb = k >> 5, kk = k & 31;
  int nb = n >> 4, nn = n & 15;
  int lane = nn + (((kk >> 3) & 1) << 4);
  int j    = ((kk & 7) >> 1) + (((kk >> 4) & 1) << 2);
  int slot = (lane << 4) + (j << 1) + (kk & 1);
  size_t tile = (size_t)nb * (HIDD >> 5) + kb;
  P[tile * 512 + slot] = (__bf16)W[idx];
}

// ---------------------------------------------------------------------------
// Tiny first layers of the MLP encoders (K=2 and K=1): pure streaming.
// ---------------------------------------------------------------------------
__global__ void node_l1(const float* __restrict__ x, const float* __restrict__ w0,
                        const float* __restrict__ b0, __bf16* __restrict__ out) {
  int i = blockIdx.x * blockDim.x + threadIdx.x;
  if (i >= NNODES * HIDD) return;
  int n = i >> 8, j = i & 255;
  float v = x[n * 2 + 0] * w0[j] + x[n * 2 + 1] * w0[HIDD + j] + b0[j];
  out[i] = (__bf16)(v > 0.0f ? v : 0.0f);
}

__global__ void edge_l1(const float* __restrict__ ea, const float* __restrict__ w0,
                        const float* __restrict__ b0, __bf16* __restrict__ out) {
  int i = blockIdx.x * blockDim.x + threadIdx.x;
  if (i >= NEDGES * HIDD) return;
  int e = i >> 8, j = i & 255;
  float v = ea[e] * w0[j] + b0[j];
  out[i] = (__bf16)(v > 0.0f ? v : 0.0f);
}

// ---------------------------------------------------------------------------
// GEMM: C[M,ncols] = A[M,256](bf16) * W(packed bf16) + bias, K=256.
// Block: TDM-stage the whole packed weight into LDS once, then 8 waves x
// 4 reps, each wave computing a 16x64 strip (A fragment reused over 4 tiles).
// ---------------------------------------------------------------------------
__global__ __launch_bounds__(256) void gemm16x64_bf16(
    const __bf16* __restrict__ A, const __bf16* __restrict__ Bp,
    const float* __restrict__ bias, float* __restrict__ outF,
    __bf16* __restrict__ outB, int M, int ncols) {
  extern __shared__ __bf16 smem[];
  const int K = HIDD;
  const int wid   = threadIdx.x >> 5;
  const int lane  = threadIdx.x & 31;
  const int strips = ncols >> 6;
  const int kTiles = K >> 5;

  if (threadIdx.x < 32) {                       // wave 0 issues the TDM copy
    tdm_load_1d(0u, Bp, (unsigned)(ncols * K * 2));
    __builtin_amdgcn_s_wait_tensorcnt(0);
  }
  __syncthreads();

  const int ksel = lane >> 4;
  const int hi   = lane >> 4;

  for (int rep = 0; rep < 4; ++rep) {
    const int wflat   = (blockIdx.x * 4 + rep) * 8 + wid;
    const int rowTile = wflat / strips;
    const int col64   = (wflat - rowTile * strips) << 6;
    if (rowTile * 16 >= M) continue;            // wave-uniform; grids exact

    const int mrow = rowTile * 16 + (lane & 15);
    const __bf16* arow = A + (size_t)mrow * K + ksel * 8;

    v8f acc[4] = {};
    for (int k0 = 0; k0 < K; k0 += 32) {
      union { v16bf v; u32x4 q[2]; } a;
      a.q[0] = *(const u32x4*)(arow + k0);
      a.q[1] = *(const u32x4*)(arow + k0 + 16);
#pragma unroll
      for (int t = 0; t < 4; ++t) {
        const __bf16* bp = smem +
            ((size_t)((col64 >> 4) + t) * kTiles + (k0 >> 5)) * 512 + lane * 16;
        v16bf b = *(const v16bf*)bp;            // ds_load_b128 x2
        acc[t] = __builtin_amdgcn_wmma_f32_16x16x32_bf16(
            false, a.v, false, b, (short)0, acc[t], false, false);
      }
    }

    const int cbase = col64 + (lane & 15);
#pragma unroll
    for (int t = 0; t < 4; ++t) {
#pragma unroll
      for (int r = 0; r < 8; ++r) {
        int rr = rowTile * 16 + hi * 8 + r;     // C layout: M = r + 8*hi
        int cc = cbase + t * 16;
        float v = acc[t][r];
        if (bias) v += bias[cc];
        if (outF) outF[(size_t)rr * ncols + cc] = v;
        if (outB) outB[(size_t)rr * ncols + cc] = (__bf16)v;
      }
    }
  }
}

// ---------------------------------------------------------------------------
// Fused pass A: per wave, 16 edges. We (packed) staged in LDS via TDM.
// For each 16-col tile: ee = e @ We (8x WMMA), z = ee + xl[src] + xr[dst]
// (gathered in C layout), leaky-ReLU, dot with att via 16-lane butterflies;
// per head write logits and atomic-max m[dst].
// ---------------------------------------------------------------------------
__global__ __launch_bounds__(256) void gat_edge_logits(
    const __bf16* __restrict__ ebf, const __bf16* __restrict__ WeP,
    const float* __restrict__ att,      // 256 floats (this layer, [8][32] flat)
    const float* __restrict__ xl, const float* __restrict__ xr,
    const int* __restrict__ src, const int* __restrict__ dst,
    float* __restrict__ logits, float* __restrict__ m) {
  extern __shared__ __bf16 smem[];
  const int K = HIDD;
  const int wid  = threadIdx.x >> 5;
  const int lane = threadIdx.x & 31;
  const int hi = lane >> 4;
  const int r0 = hi * 8;
  const int kTiles = K >> 5;

  if (threadIdx.x < 32) {
    tdm_load_1d(0u, WeP, (unsigned)(HIDD * K * 2));
    __builtin_amdgcn_s_wait_tensorcnt(0);
  }
  __syncthreads();

  for (int rep = 0; rep < 4; ++rep) {
    const int wflat = (blockIdx.x * 4 + rep) * 8 + wid;
    const int ebase = wflat * 16;

    int srow[8], drow[8];
#pragma unroll
    for (int r = 0; r < 8; ++r) {
      srow[r] = src[ebase + r0 + r];
      drow[r] = dst[ebase + r0 + r];
    }

    const int erow = ebase + (lane & 15);
    const __bf16* arow = ebf + (size_t)erow * K + hi * 8;

    float lacc[8];
#pragma unroll
    for (int r = 0; r < 8; ++r) lacc[r] = 0.0f;

    for (int n0 = 0; n0 < HIDD; n0 += 16) {
      v8f c = {};
#pragma unroll
      for (int k0 = 0; k0 < K; k0 += 32) {
        union { v16bf v; u32x4 q[2]; } a;
        a.q[0] = *(const u32x4*)(arow + k0);
        a.q[1] = *(const u32x4*)(arow + k0 + 16);
        const __bf16* bp = smem +
            ((size_t)(n0 >> 4) * kTiles + (k0 >> 5)) * 512 + lane * 16;
        v16bf b = *(const v16bf*)bp;
        c = __builtin_amdgcn_wmma_f32_16x16x32_bf16(
            false, a.v, false, b, (short)0, c, false, false);
      }
      const int col = n0 + (lane & 15);
      const float av = att[col];
#pragma unroll
      for (int r = 0; r < 8; ++r) {
        float z = c[r] + xl[(size_t)srow[r] * HIDD + col]
                       + xr[(size_t)drow[r] * HIDD + col];
        z = z > 0.0f ? z : NEG_SLOPE * z;
        lacc[r] += z * av;
      }
      if (n0 & 16) {                       // finished head n0>>5
        const int head = n0 >> 5;
#pragma unroll
        for (int r = 0; r < 8; ++r) {
          float v = lacc[r];
          v += __shfl_xor(v, 1);
          v += __shfl_xor(v, 2);
          v += __shfl_xor(v, 4);
          v += __shfl_xor(v, 8);           // reduce within each 16-lane half
          if ((lane & 15) == 0) {
            logits[(size_t)(ebase + r0 + r) * NHEAD + head] = v;
            atomicMaxF32(&m[(size_t)drow[r] * NHEAD + head], v);
          }
          lacc[r] = 0.0f;
        }
      }
    }
  }
}

// ---------------------------------------------------------------------------
// Softmax pieces + weighted scatter + node update.
// ---------------------------------------------------------------------------
__global__ void init_ms(float* __restrict__ m, float* __restrict__ s,
                        float* __restrict__ acc) {
  int i = blockIdx.x * blockDim.x + threadIdx.x;
  if (i < NNODES * NHEAD) { m[i] = -INFINITY; s[i] = 0.0f; }
  if (i < NNODES * HIDD)  acc[i] = 0.0f;
}

__global__ void seg_exp_sum(const float* __restrict__ logits,
                            const int* __restrict__ dst,
                            const float* __restrict__ m,
                            float* __restrict__ ex, float* __restrict__ s) {
  int i = blockIdx.x * blockDim.x + threadIdx.x;
  if (i >= NEDGES * NHEAD) return;
  int e = i >> 3, h = i & 7;
  int d = dst[e];
  float v = __expf(logits[i] - m[d * NHEAD + h]);
  ex[i] = v;
  atomicAdd(&s[d * NHEAD + h], v);
}

__global__ void scatter_out(const float* __restrict__ ex,
                            const float* __restrict__ s,
                            const float* __restrict__ xl,
                            const int* __restrict__ src,
                            const int* __restrict__ dst,
                            float* __restrict__ acc) {
  int i = blockIdx.x * blockDim.x + threadIdx.x;   // E * 64 threads, 4 ch each
  if (i >= NEDGES * 64) return;
  int e = i >> 6, q = i & 63;
  int col = q << 2;
  int h = col >> 5;
  int d = dst[e];
  float alpha = ex[e * NHEAD + h] / (s[d * NHEAD + h] + 1e-16f);
  f32x4 xv = *(const f32x4*)(xl + (size_t)src[e] * HIDD + col);
  float* ap = acc + (size_t)d * HIDD + col;
  atomicAdd(ap + 0, xv.x * alpha);
  atomicAdd(ap + 1, xv.y * alpha);
  atomicAdd(ap + 2, xv.z * alpha);
  atomicAdd(ap + 3, xv.w * alpha);
}

__global__ void node_update(const float* __restrict__ acc,
                            const float* __restrict__ resid,   // h@Wres + bias
                            __bf16* __restrict__ hbf) {
  int i = blockIdx.x * blockDim.x + threadIdx.x;
  if (i >= NNODES * HIDD) return;
  float v = acc[i] + resid[i];
  hbf[i] = (__bf16)(v > 0.0f ? v : 0.0f);
}

// ---------------------------------------------------------------------------
// Host side
// ---------------------------------------------------------------------------
extern "C" void kernel_launch(void* const* d_in, const int* in_sizes, int n_in,
                              void* d_out, int out_size, void* d_ws, size_t ws_size,
                              hipStream_t stream) {
  (void)in_sizes; (void)n_in; (void)out_size; (void)ws_size;
  const float* x        = (const float*)d_in[0];
  const int*   eidx     = (const int*)  d_in[1];
  const float* eattr    = (const float*)d_in[2];
  /* batch d_in[3] unused: equal-size graphs -> dense batch is a reshape */
  const float* node_w0  = (const float*)d_in[4];
  const float* node_b0  = (const float*)d_in[5];
  const float* node_w1  = (const float*)d_in[6];
  const float* node_b1  = (const float*)d_in[7];
  const float* edge_w0  = (const float*)d_in[8];
  const float* edge_b0  = (const float*)d_in[9];
  const float* edge_w1  = (const float*)d_in[10];
  const float* edge_b1  = (const float*)d_in[11];
  const float* Wl       = (const float*)d_in[12];
  const float* bl       = (const float*)d_in[13];
  const float* Wr       = (const float*)d_in[14];
  const float* br       = (const float*)d_in[15];
  const float* We       = (const float*)d_in[16];
  const float* att      = (const float*)d_in[17];
  const float* bias     = (const float*)d_in[18];
  const float* Wres     = (const float*)d_in[19];
  const float* lin_w    = (const float*)d_in[20];
  const float* lin_b    = (const float*)d_in[21];
  const int* src = eidx;
  const int* dstp = eidx + NEDGES;

  // ---- workspace carve-up (256B aligned) ----
  size_t off = 0;
  auto carve = [&](size_t bytes) -> char* {
    off = (off + 255) & ~(size_t)255;
    char* p = (char*)d_ws + off;
    off += bytes;
    return p;
  };
  const size_t PK = (size_t)HIDD * HIDD * 2;          // 128 KB packed weight
  __bf16* pNodeW1 = (__bf16*)carve(PK);
  __bf16* pEdgeW1 = (__bf16*)carve(PK);
  __bf16* pWl[2]  = {(__bf16*)carve(PK), (__bf16*)carve(PK)};
  __bf16* pWr[2]  = {(__bf16*)carve(PK), (__bf16*)carve(PK)};
  __bf16* pWe[2]  = {(__bf16*)carve(PK), (__bf16*)carve(PK)};
  __bf16* pWres[2]= {(__bf16*)carve(PK), (__bf16*)carve(PK)};
  __bf16* pLin    = (__bf16*)carve((size_t)HIDD * OUTD * 2);
  __bf16* hbf     = (__bf16*)carve((size_t)NNODES * HIDD * 2);
  __bf16* ebf     = (__bf16*)carve((size_t)NEDGES * HIDD * 2);
  float*  logits  = (float*) carve((size_t)NEDGES * NHEAD * 4);
  float*  ex      = (float*) carve((size_t)NEDGES * NHEAD * 4);
  float*  mbuf    = (float*) carve((size_t)NNODES * NHEAD * 4);
  float*  sbuf    = (float*) carve((size_t)NNODES * NHEAD * 4);
  // Big region: first bf16 encoder temp [E,256], later 4x f32 [N,256].
  char*   T       = carve((size_t)NEDGES * HIDD * 2);
  __bf16* tmp_bf  = (__bf16*)T;
  float*  xl      = (float*)(T);
  float*  xr      = (float*)(T + (size_t)NNODES * HIDD * 4);
  float*  resid   = (float*)(T + (size_t)NNODES * HIDD * 8);
  float*  acc     = (float*)(T + (size_t)NNODES * HIDD * 12);

  auto repack = [&](const float* W, __bf16* P, int ncols) {
    int tot = HIDD * ncols;
    repack_w<<<(tot + 255) / 256, 256, 0, stream>>>(W, P, ncols);
  };
  auto gemm = [&](const __bf16* A, const __bf16* Bp, const float* b,
                  float* oF, __bf16* oB, int M, int ncols) {
    int blocks = (M / 16) * (ncols / 64) / 32;        // 8 waves x 4 reps
    size_t smemB = (size_t)ncols * HIDD * 2;          // whole packed weight
    gemm16x64_bf16<<<blocks, 256, smemB, stream>>>(A, Bp, b, oF, oB, M, ncols);
  };

  // ---- repack all weights into WMMA B-fragment layout ----
  repack(node_w1, pNodeW1, HIDD);
  repack(edge_w1, pEdgeW1, HIDD);
  for (int l = 0; l < 2; ++l) {
    size_t wo = (size_t)l * HIDD * HIDD;
    repack(Wl + wo,   pWl[l],   HIDD);
    repack(Wr + wo,   pWr[l],   HIDD);
    repack(We + wo,   pWe[l],   HIDD);
    repack(Wres + wo, pWres[l], HIDD);
  }
  repack(lin_w, pLin, OUTD);

  // ---- encoders ----
  node_l1<<<NNODES * HIDD / 256, 256, 0, stream>>>(x, node_w0, node_b0, tmp_bf);
  gemm(tmp_bf, pNodeW1, node_b1, nullptr, hbf, NNODES, HIDD);     // h -> bf16
  edge_l1<<<NEDGES * HIDD / 256, 256, 0, stream>>>(eattr, edge_w0, edge_b0, tmp_bf);
  gemm(tmp_bf, pEdgeW1, edge_b1, nullptr, ebf, NEDGES, HIDD);     // e -> bf16

  // ---- two GATv2 layers ----
  for (int l = 0; l < 2; ++l) {
    gemm(hbf, pWl[l],   bl + l * HIDD,   xl,    nullptr, NNODES, HIDD);
    gemm(hbf, pWr[l],   br + l * HIDD,   xr,    nullptr, NNODES, HIDD);
    gemm(hbf, pWres[l], bias + l * HIDD, resid, nullptr, NNODES, HIDD);
    init_ms<<<NNODES * HIDD / 256, 256, 0, stream>>>(mbuf, sbuf, acc);
    gat_edge_logits<<<NEDGES / 16 / 8 / 4, 256, (size_t)HIDD * HIDD * 2, stream>>>(
        ebf, pWe[l], att + l * NHEAD * 32, xl, xr, src, dstp, logits, mbuf);
    seg_exp_sum<<<NEDGES * NHEAD / 256, 256, 0, stream>>>(logits, dstp, mbuf, ex, sbuf);
    scatter_out<<<NEDGES * 64 / 256, 256, 0, stream>>>(ex, sbuf, xl, src, dstp, acc);
    node_update<<<NNODES * HIDD / 256, 256, 0, stream>>>(acc, resid, hbf);
  }

  // ---- final linear -> d_out ([16,1024,128] == row-major [N,128]) ----
  gemm(hbf, pLin, lin_b, (float*)d_out, nullptr, NNODES, OUTD);
}
